// TransformerBlock_56556129354398
// MI455X (gfx1250) — matrix operand order repaired and checked by reading, not compiled, not run on hardware
//
#include <hip/hip_runtime.h>
#include <math.h>

// ---------------------------------------------------------------------------
// MI455X / gfx1250 transformer block, bf16 WMMA path (v_wmma_f32_16x16x32_bf16)
// + Tensor Data Mover staging for GEMM A-tiles (tensor_load_to_lds)
// ---------------------------------------------------------------------------

typedef __attribute__((ext_vector_type(16))) __bf16 v16bf;
typedef __attribute__((ext_vector_type(8)))  float  v8f;

typedef unsigned int uint32x4_t __attribute__((ext_vector_type(4)));
typedef int          int32x8_t  __attribute__((ext_vector_type(8)));
typedef int          int32x4_t  __attribute__((ext_vector_type(4)));

#if __has_builtin(__builtin_amdgcn_tensor_load_to_lds) && \
    __has_builtin(__builtin_amdgcn_s_wait_tensorcnt)
#define HAVE_TDM 1
#endif

#define N_EMBD   1024
#define N_HEAD   16
#define HEAD_DIM 64
#define TSEQ     2048
#define NBATCH   4
#define MROWS    (NBATCH * TSEQ)   // 8192

union AFrag {
    v16bf v;
    uint4 u[2];
    __bf16 h[16];
};

// ---------------------------------------------------------------------------
// f32 -> bf16 elementwise convert (weights)
// ---------------------------------------------------------------------------
__global__ void f32_to_bf16_k(const float* __restrict__ in,
                              __bf16* __restrict__ out, int n) {
    int i = blockIdx.x * blockDim.x + threadIdx.x;
    int stride = gridDim.x * blockDim.x;
    for (; i < n; i += stride) out[i] = (__bf16)in[i];
}

// ---------------------------------------------------------------------------
// LayerNorm over last dim (1024), f32 in -> bf16 out. One block per row.
// ---------------------------------------------------------------------------
__global__ __launch_bounds__(256) void layernorm_k(
    const float* __restrict__ x, const float* __restrict__ w,
    const float* __restrict__ b, __bf16* __restrict__ out) {
    __shared__ float red[16];
    const int row = blockIdx.x;
    const int tid = threadIdx.x;
    const float* xr = x + (size_t)row * N_EMBD;

    float v[4];
    float s = 0.f, ss = 0.f;
#pragma unroll
    for (int i = 0; i < 4; ++i) {
        v[i] = xr[tid + i * 256];
        s += v[i];
        ss += v[i] * v[i];
    }
#pragma unroll
    for (int m = 16; m >= 1; m >>= 1) {
        s  += __shfl_xor(s, m, 32);
        ss += __shfl_xor(ss, m, 32);
    }
    const int wid = tid >> 5;
    if ((tid & 31) == 0) { red[wid] = s; red[8 + wid] = ss; }
    __syncthreads();
    float ts = 0.f, tss = 0.f;
#pragma unroll
    for (int j = 0; j < 8; ++j) { ts += red[j]; tss += red[8 + j]; }
    const float mean = ts * (1.f / N_EMBD);
    const float var  = tss * (1.f / N_EMBD) - mean * mean;
    const float rstd = rsqrtf(var + 1e-6f);

    __bf16* orow = out + (size_t)row * N_EMBD;
#pragma unroll
    for (int i = 0; i < 4; ++i) {
        int c = tid + i * 256;
        orow[c] = (__bf16)(w[c] * ((v[i] - mean) * rstd) + b[c]);
    }
}

// ---------------------------------------------------------------------------
// Tiled bf16 GEMM with f32 accumulation.
//   C[M,N] = A[M,K] * B[K,N] + bias  (+ epilogue)
// Block: 256 threads (8 waves). Tile 128x128x32. Wave tile 32x64 (2x4 WMMA).
// EPI: 0 = bf16 out (bias)     1 = f32 out (bias + residual)
//      2 = bf16 out (bias + exact GELU)
// A-tile staged via Tensor Data Mover (one tensor_load_to_lds per K step,
// issued by wave 0, synced with s_wait_tensorcnt) when available.
// ---------------------------------------------------------------------------
template <int EPI>
__global__ __launch_bounds__(256) void gemm_bf16_k(
    const __bf16* __restrict__ A, const __bf16* __restrict__ Bm,
    const float* __restrict__ bias, const float* __restrict__ res,
    float* __restrict__ outF, __bf16* __restrict__ outH,
    int M, int N, int K) {
    __shared__ __bf16 As[128][32];    // A tile, row-major
    __shared__ __bf16 BTs[128][32];   // B tile, transposed: BTs[n][k]

    const int tid  = threadIdx.x;
    const int lane = tid & 31;
    const int wave = tid >> 5;
    const int lh   = lane >> 4;       // lane half (0/1)
    const int nlo  = lane & 15;
    const int rowBase = blockIdx.y * 128;
    const int colBase = blockIdx.x * 128;
    const int wRow = (wave & 3) * 32;
    const int wCol = (wave >> 2) * 64;

    v8f acc[2][4];
#pragma unroll
    for (int i = 0; i < 2; ++i)
#pragma unroll
        for (int j = 0; j < 4; ++j)
#pragma unroll
            for (int e = 0; e < 8; ++e) acc[i][j][e] = 0.f;

#ifdef HAVE_TDM
    // D# group 1..3 are loop-invariant (tensor = A[M,K] bf16, tile = 128x32).
    int32x8_t g1;
    g1[0] = (1 << 16);                         // wg_mask=0, data_size=1 (2B)
    g1[1] = (K & 0xFFFF) << 16;                // tensor_dim0[15:0]  (bits 63:48)
    g1[2] = ((unsigned)K >> 16) | ((M & 0xFFFF) << 16);   // dim0 hi / dim1 lo
    g1[3] = ((unsigned)M >> 16) | (32 << 16);  // dim1 hi / tile_dim0 = 32
    g1[4] = 128;                               // tile_dim1 = 128, tile_dim2 = 0
    g1[5] = K;                                 // tensor_dim0_stride low 32
    g1[6] = 0;                                 // stride hi / dim1_stride lo
    g1[7] = 0;
    int32x4_t gz4;
    gz4[0] = 0; gz4[1] = 0; gz4[2] = 0; gz4[3] = 0;
    int32x8_t gz8;
#pragma unroll
    for (int i = 0; i < 8; ++i) gz8[i] = 0;
    const unsigned ldsA = (unsigned)(size_t)(&As[0][0]);
#endif

    for (int k0 = 0; k0 < K; k0 += 32) {
        // ---- stage A tile (128x32 bf16)
#ifdef HAVE_TDM
        if (tid < 32) {                        // wave 0 drives the TDM
            unsigned long long ga =
                (unsigned long long)(size_t)(A + (size_t)rowBase * K + k0);
            uint32x4_t g0;
            g0[0] = 1u;                                    // count=1, user D#
            g0[1] = ldsA;                                  // lds_addr
            g0[2] = (unsigned)(ga & 0xFFFFFFFFu);          // global_addr lo
            g0[3] = (unsigned)((ga >> 32) & 0x01FFFFFFu) | (2u << 30); // type=2
            __builtin_amdgcn_tensor_load_to_lds(g0, g1, gz4, gz4, gz8, 0);
            __builtin_amdgcn_s_wait_tensorcnt(0);
        }
#else
#pragma unroll
        for (int i = 0; i < 2; ++i) {
            int c  = tid + i * 256;       // 512 chunks of 8 bf16
            int r  = c >> 2;
            int c8 = (c & 3) << 3;
            const __bf16* src = A + (size_t)(rowBase + r) * K + k0 + c8;
            *(uint4*)&As[r][c8] = *(const uint4*)src;
            __builtin_prefetch(src + 32, 0, 1);
        }
#endif
        // ---- stage B tile transposed (BTs[n][k])
#pragma unroll
        for (int i = 0; i < 2; ++i) {
            int c  = tid + i * 256;
            int kr = c >> 4;
            int nc = (c & 15) << 3;
            const __bf16* src = Bm + (size_t)(k0 + kr) * N + colBase + nc;
            union { uint4 u; __bf16 h[8]; } t;
            t.u = *(const uint4*)src;
            __builtin_prefetch(src + 32 * (size_t)N, 0, 1);
#pragma unroll
            for (int j = 0; j < 8; ++j) BTs[nc + j][kr] = t.h[j];
        }
        __syncthreads();

        // ---- fragments from LDS (all 16B ds_load_b128)
        AFrag a[2], bfr[4];
#pragma unroll
        for (int mt = 0; mt < 2; ++mt) {
            int r = wRow + mt * 16 + nlo;
            a[mt].u[0] = *(const uint4*)&As[r][lh * 8];        // K = 8*lh+0..7
            a[mt].u[1] = *(const uint4*)&As[r][16 + lh * 8];   // K = 16+8*lh+..
        }
#pragma unroll
        for (int nt = 0; nt < 4; ++nt) {
            int cc = wCol + nt * 16 + nlo;
            bfr[nt].u[0] = *(const uint4*)&BTs[cc][lh * 8];      // K = 8*lh+0..7
            bfr[nt].u[1] = *(const uint4*)&BTs[cc][16 + lh * 8]; // K = 16+8*lh+..
        }
#pragma unroll
        for (int mt = 0; mt < 2; ++mt)
#pragma unroll
            for (int nt = 0; nt < 4; ++nt)
                acc[mt][nt] = __builtin_amdgcn_wmma_f32_16x16x32_bf16(
                    false, a[mt].v, false, bfr[nt].v, (short)0, acc[mt][nt],
                    false, false);
        __syncthreads();
    }

    // ---- epilogue (C/D layout: row = r + 8*lh, col = lane&15)
#pragma unroll
    for (int mt = 0; mt < 2; ++mt) {
#pragma unroll
        for (int nt = 0; nt < 4; ++nt) {
            int gcol = colBase + wCol + nt * 16 + nlo;
            float bv = bias[gcol];
#pragma unroll
            for (int r = 0; r < 8; ++r) {
                int grow = rowBase + wRow + mt * 16 + r + 8 * lh;
                size_t idx = (size_t)grow * N + gcol;
                float v = acc[mt][nt][r] + bv;
                if (EPI == 1) {
                    outF[idx] = v + res[idx];
                } else if (EPI == 2) {
                    outH[idx] = (__bf16)(0.5f * v * (1.f + erff(v * 0.70710678118f)));
                } else {
                    outH[idx] = (__bf16)v;
                }
            }
        }
    }
}

// ---------------------------------------------------------------------------
// WMMA flash attention. One wave (32 threads) per (b, h, 16-query tile).
// qkv: bf16 [8192][3072] (q | k | v, each head 64 wide). y: bf16 [8192][1024].
// ---------------------------------------------------------------------------
__global__ __launch_bounds__(32) void attn_k(const __bf16* __restrict__ qkv,
                                             __bf16* __restrict__ y) {
    __shared__ __bf16 Ps[16][32];   // P tile scratch for C->A relayout

    const int lane = threadIdx.x;
    const int lh   = lane >> 4;
    const int nlo  = lane & 15;
    const int qt = blockIdx.x & 127;          // T/16 = 128 query tiles
    const int bh = blockIdx.x >> 7;
    const int b  = bh >> 4;
    const int h  = bh & 15;
    const int q0 = qt * 16;
    const size_t rowB = (size_t)b * TSEQ;
    const int qoff = h * HEAD_DIM;
    const int koff = qoff + N_EMBD;
    const int voff = qoff + 2 * N_EMBD;
    const int LD = 3 * N_EMBD;

    // Q fragments (A layout, 16 queries x 64 dims -> 2 K-steps of 32)
    AFrag qf[2];
    {
        const __bf16* qrow = qkv + (rowB + q0 + nlo) * LD + qoff;
#pragma unroll
        for (int ks = 0; ks < 2; ++ks) {
            qf[ks].u[0] = *(const uint4*)(qrow + ks * 32 + lh * 8);
            qf[ks].u[1] = *(const uint4*)(qrow + ks * 32 + 16 + lh * 8);
        }
    }

    v8f O[4];
#pragma unroll
    for (int dt = 0; dt < 4; ++dt)
#pragma unroll
        for (int e = 0; e < 8; ++e) O[dt][e] = 0.f;
    float mrow[8], lrow[8];
#pragma unroll
    for (int r = 0; r < 8; ++r) { mrow[r] = -INFINITY; lrow[r] = 0.f; }

    const int nkt = (q0 + 47) >> 5;   // causal: key tiles covering keys <= q0+15
    for (int kt = 0; kt < nkt; ++kt) {
        const int kb = kt * 32;

        // ---- S = Q * K^T  (two 16x16 key sub-tiles, 2 K-steps of d each)
        v8f s[2];
#pragma unroll
        for (int st = 0; st < 2; ++st) {
            const __bf16* krow = qkv + (rowB + kb + st * 16 + nlo) * LD + koff;
            AFrag kf0, kf1;                   // B layout: lane = key column
            kf0.u[0] = *(const uint4*)(krow + lh * 8);
            kf0.u[1] = *(const uint4*)(krow + 16 + lh * 8);
            kf1.u[0] = *(const uint4*)(krow + 32 + lh * 8);
            kf1.u[1] = *(const uint4*)(krow + 32 + 16 + lh * 8);
            v8f z;
#pragma unroll
            for (int e = 0; e < 8; ++e) z[e] = 0.f;
            z = __builtin_amdgcn_wmma_f32_16x16x32_bf16(
                false, qf[0].v, false, kf0.v, (short)0, z, false, false);
            s[st] = __builtin_amdgcn_wmma_f32_16x16x32_bf16(
                false, qf[1].v, false, kf1.v, (short)0, z, false, false);
        }

        // ---- scale + causal mask + online softmax (f32)
        float p[2][8];
#pragma unroll
        for (int r = 0; r < 8; ++r) {
            int qrow = q0 + r + 8 * lh;
            float s0 = s[0][r] * 0.125f;          // 1/sqrt(64)
            float s1 = s[1][r] * 0.125f;
            if (kb + nlo > qrow)      s0 = -INFINITY;
            if (kb + 16 + nlo > qrow) s1 = -INFINITY;
            float m = fmaxf(s0, s1);
            m = fmaxf(m, __shfl_xor(m, 1, 32));
            m = fmaxf(m, __shfl_xor(m, 2, 32));
            m = fmaxf(m, __shfl_xor(m, 4, 32));
            m = fmaxf(m, __shfl_xor(m, 8, 32));
            float mnew = fmaxf(mrow[r], m);
            float p0 = __expf(s0 - mnew);
            float p1 = __expf(s1 - mnew);
            p[0][r] = p0; p[1][r] = p1;
            float ps = p0 + p1;
            ps += __shfl_xor(ps, 1, 32);
            ps += __shfl_xor(ps, 2, 32);
            ps += __shfl_xor(ps, 4, 32);
            ps += __shfl_xor(ps, 8, 32);
            float alpha = __expf(mrow[r] - mnew);
            lrow[r] = lrow[r] * alpha + ps;
            mrow[r] = mnew;
#pragma unroll
            for (int dt = 0; dt < 4; ++dt) O[dt][r] *= alpha;
        }

        // ---- P (C layout f32) -> LDS -> A layout bf16
#pragma unroll
        for (int r = 0; r < 8; ++r) {
            Ps[r + 8 * lh][nlo]      = (__bf16)p[0][r];
            Ps[r + 8 * lh][16 + nlo] = (__bf16)p[1][r];
        }
        __syncthreads();   // single-wave workgroup: barrier is S_NOP, fence remains
        AFrag pf;
        pf.u[0] = *(const uint4*)&Ps[nlo][lh * 8];
        pf.u[1] = *(const uint4*)&Ps[nlo][16 + lh * 8];

        // ---- O += P * V  (4 output d-tiles of 16)
#pragma unroll
        for (int dt = 0; dt < 4; ++dt) {
            AFrag vf;
            int d = dt * 16 + nlo;
#pragma unroll
            for (int j = 0; j < 8; ++j) {
                int key = kb + 8 * lh + 2 * j + 8 * (j >> 2);
                vf.h[2 * j]     = qkv[(rowB + key) * LD + voff + d];
                vf.h[2 * j + 1] = qkv[(rowB + key + 1) * LD + voff + d];
            }
            O[dt] = __builtin_amdgcn_wmma_f32_16x16x32_bf16(
                false, pf.v, false, vf.v, (short)0, O[dt], false, false);
        }
        __syncthreads();
    }

    // ---- normalize and store y (bf16)
#pragma unroll
    for (int r = 0; r < 8; ++r) {
        float inv = 1.f / lrow[r];
        int t = q0 + r + 8 * lh;
#pragma unroll
        for (int dt = 0; dt < 4; ++dt) {
            y[(rowB + t) * N_EMBD + qoff + dt * 16 + nlo] =
                (__bf16)(O[dt][r] * inv);
        }
    }
}

// ---------------------------------------------------------------------------
// Launcher
// ---------------------------------------------------------------------------
extern "C" void kernel_launch(void* const* d_in, const int* in_sizes, int n_in,
                              void* d_out, int out_size, void* d_ws, size_t ws_size,
                              hipStream_t stream) {
    const float* x      = (const float*)d_in[0];
    const float* ln1_w  = (const float*)d_in[1];
    const float* ln1_b  = (const float*)d_in[2];
    const float* ln2_w  = (const float*)d_in[3];
    const float* ln2_b  = (const float*)d_in[4];
    const float* w_attn = (const float*)d_in[5];
    const float* b_attn = (const float*)d_in[6];
    const float* w_proj = (const float*)d_in[7];
    const float* b_proj = (const float*)d_in[8];
    const float* w_fc   = (const float*)d_in[9];
    const float* b_fc   = (const float*)d_in[10];
    const float* w_fc2  = (const float*)d_in[11];
    const float* b_fc2  = (const float*)d_in[12];

    char* ws = (char*)d_ws;
    size_t off = 0;
    auto alloc = [&](size_t bytes) -> void* {
        void* p = ws + off;
        off += (bytes + 255) & ~(size_t)255;
        return p;
    };

    __bf16* wattn_h = (__bf16*)alloc((size_t)N_EMBD * 3 * N_EMBD * 2);
    __bf16* wproj_h = (__bf16*)alloc((size_t)N_EMBD * N_EMBD * 2);
    __bf16* wfc_h   = (__bf16*)alloc((size_t)N_EMBD * 4 * N_EMBD * 2);
    __bf16* wfc2_h  = (__bf16*)alloc((size_t)4 * N_EMBD * N_EMBD * 2);
    __bf16* ln1_h   = (__bf16*)alloc((size_t)MROWS * N_EMBD * 2);
    __bf16* qkv_h   = (__bf16*)alloc((size_t)MROWS * 3 * N_EMBD * 2);
    __bf16* y_h     = (__bf16*)alloc((size_t)MROWS * N_EMBD * 2);
    float*  x1_f    = (float*) alloc((size_t)MROWS * N_EMBD * 4);
    __bf16* ln2_h   = (__bf16*)alloc((size_t)MROWS * N_EMBD * 2);
    __bf16* hfc_h   = (__bf16*)alloc((size_t)MROWS * 4 * N_EMBD * 2);

    // 1. weights -> bf16
    f32_to_bf16_k<<<512, 256, 0, stream>>>(w_attn, wattn_h, N_EMBD * 3 * N_EMBD);
    f32_to_bf16_k<<<512, 256, 0, stream>>>(w_proj, wproj_h, N_EMBD * N_EMBD);
    f32_to_bf16_k<<<512, 256, 0, stream>>>(w_fc,   wfc_h,   N_EMBD * 4 * N_EMBD);
    f32_to_bf16_k<<<512, 256, 0, stream>>>(w_fc2,  wfc2_h,  4 * N_EMBD * N_EMBD);

    // 2. LN1
    layernorm_k<<<MROWS, 256, 0, stream>>>(x, ln1_w, ln1_b, ln1_h);

    // 3. QKV GEMM: [8192,1024] x [1024,3072] -> bf16
    gemm_bf16_k<0><<<dim3(3 * N_EMBD / 128, MROWS / 128), 256, 0, stream>>>(
        ln1_h, wattn_h, b_attn, nullptr, nullptr, qkv_h,
        MROWS, 3 * N_EMBD, N_EMBD);

    // 4. flash attention (WMMA)
    attn_k<<<NBATCH * N_HEAD * (TSEQ / 16), 32, 0, stream>>>(qkv_h, y_h);

    // 5. proj GEMM + residual: x1 = x + y*Wp + bp (f32)
    gemm_bf16_k<1><<<dim3(N_EMBD / 128, MROWS / 128), 256, 0, stream>>>(
        y_h, wproj_h, b_proj, x, x1_f, nullptr,
        MROWS, N_EMBD, N_EMBD);

    // 6. LN2
    layernorm_k<<<MROWS, 256, 0, stream>>>(x1_f, ln2_w, ln2_b, ln2_h);

    // 7. FC GEMM + exact GELU -> bf16
    gemm_bf16_k<2><<<dim3(4 * N_EMBD / 128, MROWS / 128), 256, 0, stream>>>(
        ln2_h, wfc_h, b_fc, nullptr, nullptr, hfc_h,
        MROWS, 4 * N_EMBD, N_EMBD);

    // 8. FC2 GEMM + residual -> d_out (f32)
    gemm_bf16_k<1><<<dim3(N_EMBD / 128, MROWS / 128), 256, 0, stream>>>(
        hfc_h, wfc2_h, b_fc2, x1_f, (float*)d_out, nullptr,
        MROWS, N_EMBD, 4 * N_EMBD);
}
